// Layer154_39290360824307
// MI455X (gfx1250) — compile-verified, matrix operand order checked by reading, not run on hardware
//
#include <hip/hip_runtime.h>

typedef __attribute__((ext_vector_type(16))) __bf16 v16bf;
typedef __attribute__((ext_vector_type(8)))  __bf16 v8bf;
typedef __attribute__((ext_vector_type(4)))  __bf16 v4bf;
typedef __attribute__((ext_vector_type(8)))  float  v8f;
typedef __attribute__((ext_vector_type(4)))  float  v4f;

#define HDIM 1024
#define NHEAD 16
#define HD 64
#define BB 4
#define SS 2048
#define MM (BB * SS)   // 8192 rows

// ---- f32 -> bf16 round-to-nearest-even -------------------------------------
__device__ __forceinline__ __bf16 f2bf(float f) {
  unsigned u = __builtin_bit_cast(unsigned, f);
  u += 0x7FFFu + ((u >> 16) & 1u);
  unsigned short h = (unsigned short)(u >> 16);
  return __builtin_bit_cast(__bf16, h);
}

// ---- WMMA bf16 16x16x32 fragment loader ------------------------------------
// A-frag: lane(0..15)=row M, lane(16..31)=row M with k-half 8..15 / 24..31.
// Elements 0..7 -> src[row][kh + 0..7], elements 8..15 -> src[row][kh + 16..23]
// B-frag uses the identical pattern with "row" = output column (source row-major
// B^T, i.e. W rows / q rows; K and V^T rows load as A). Two 16-byte loads per lane.
__device__ __forceinline__ v16bf load_frag(const __bf16* __restrict__ p, int ld) {
  const int lane = threadIdx.x & 31;
  const int r  = lane & 15;
  const int kh = (lane >> 4) * 8;
  const __bf16* q = p + (size_t)r * ld + kh;
  v8bf lo = *reinterpret_cast<const v8bf*>(q);
  v8bf hi = *reinterpret_cast<const v8bf*>(q + 16);
  v16bf f;
#pragma unroll
  for (int i = 0; i < 8; ++i) { f[i] = lo[i]; f[i + 8] = hi[i]; }
  return f;
}

#define WMMA_BF16(a, b, c) \
  __builtin_amdgcn_wmma_f32_16x16x32_bf16(false, (a), false, (b), (short)0, (c), false, false)

// ---- Kernel 0: f32 -> bf16 bulk convert ------------------------------------
__global__ void cvt_f32_to_bf16(const float* __restrict__ src,
                                __bf16* __restrict__ dst, int n) {
  int i = (blockIdx.x * blockDim.x + threadIdx.x) * 4;
  if (i + 3 < n) {
    v4f x = *reinterpret_cast<const v4f*>(src + i);
    v4bf o;
#pragma unroll
    for (int j = 0; j < 4; ++j) o[j] = f2bf(x[j]);
    *reinterpret_cast<v4bf*>(dst + i) = o;
  }
}

// ---- Kernel 1: out = X @ W^T + bias  (X:[M,H] bf16, W:[H,H] bf16 row-major) --
// Wave tile 32x64: 8 WMMAs per k-step fed by 12 b128 loads (1.5 loads/WMMA),
// ~64 accum VGPRs + 48 fragment VGPRs -> no spills. 8 waves -> 64x256 block.
// vmode=0: write [B,NH,S,HD] (q,k).  vmode=1: write [B,NH,HD,S] (v transposed).
__global__ void qkv_gemm(const __bf16* __restrict__ X, const __bf16* __restrict__ W,
                         const float* __restrict__ bias, __bf16* __restrict__ out,
                         int vmode) {
  const int wave = threadIdx.x >> 5;
  const int lane = threadIdx.x & 31;
  const int row0 = blockIdx.y * 64 + (wave & 1) * 32;
  const int col0 = blockIdx.x * 256 + (wave >> 1) * 64;

  v8f acc[2][4] = {};
  for (int k0 = 0; k0 < HDIM; k0 += 32) {
    v16bf a0 = load_frag(X + (size_t)row0 * HDIM + k0, HDIM);
    v16bf a1 = load_frag(X + (size_t)(row0 + 16) * HDIM + k0, HDIM);
    v16bf bf[4];
#pragma unroll
    for (int j = 0; j < 4; ++j)
      bf[j] = load_frag(W + (size_t)(col0 + 16 * j) * HDIM + k0, HDIM);
#pragma unroll
    for (int j = 0; j < 4; ++j) {
      acc[0][j] = WMMA_BF16(a0, bf[j], acc[0][j]);
      acc[1][j] = WMMA_BF16(a1, bf[j], acc[1][j]);
    }
  }

  const int n = lane & 15, half = lane >> 4;
#pragma unroll
  for (int mi = 0; mi < 2; ++mi) {
#pragma unroll
    for (int ni = 0; ni < 4; ++ni) {
      const int gcol = col0 + ni * 16 + n;
      const float bval = bias[gcol];
      const int h = gcol >> 6, d = gcol & 63;
#pragma unroll
      for (int r = 0; r < 8; ++r) {
        const int grow = row0 + mi * 16 + r + half * 8;   // C-layout: M = r + 8*half
        const int b = grow >> 11, s = grow & (SS - 1);
        const float val = acc[mi][ni][r] + bval;
        const size_t idx = vmode
            ? ((size_t)((b * NHEAD + h) * HD + d) * SS + s)
            : ((size_t)((b * NHEAD + h) * SS + s) * HD + d);
        out[idx] = f2bf(val);
      }
    }
  }
}

// ---- Kernel 2: flash attention ---------------------------------------------
// Scores computed transposed (S^T = K·q^T) so softmax over keys is lane-local
// (rows live in the 8 accumulator VGPRs); the p^T C-tiles repack directly into
// the B-fragment of ctx^T = V^T·p^T with zero cross-lane traffic.
__global__ void attn_fwd(const __bf16* __restrict__ Q, const __bf16* __restrict__ K,
                         const __bf16* __restrict__ V, const float* __restrict__ mask,
                         float* __restrict__ out) {
  const int wave = threadIdx.x >> 5;
  const int lane = threadIdx.x & 31;
  const int n = lane & 15, half = lane >> 4;
  const int b = blockIdx.z, hh = blockIdx.y;
  const int q0 = blockIdx.x * 64 + wave * 16;

  const size_t hOff = (size_t)(b * NHEAD + hh) * SS * HD;
  const __bf16* Qh = Q + hOff;
  const __bf16* Kh = K + hOff;
  const __bf16* Vh = V + hOff;                 // [HD, S] for this head
  const float* mrow = mask + (size_t)b * SS;

  // q^T B-fragments (reused across all key tiles)
  v16bf qb0 = load_frag(Qh + (size_t)q0 * HD, HD);
  v16bf qb1 = load_frag(Qh + (size_t)q0 * HD + 32, HD);

  v8f ctx[4] = {};                             // ctx^T: 64 d-rows x 16 q-cols
  float run_m = -1e30f, run_l = 0.f;

  for (int kt = 0; kt < SS; kt += 32) {
    // K A-fragments: 32 keys x 64 d
    v16bf ka00 = load_frag(Kh + (size_t)kt * HD, HD);
    v16bf ka01 = load_frag(Kh + (size_t)kt * HD + 32, HD);
    v16bf ka10 = load_frag(Kh + (size_t)(kt + 16) * HD, HD);
    v16bf ka11 = load_frag(Kh + (size_t)(kt + 16) * HD + 32, HD);

    if (kt + 32 < SS) {
      __builtin_prefetch(Kh + (size_t)(kt + 32) * HD, 0, 1);
      __builtin_prefetch(Vh + (kt + 32), 0, 1);
    }

    v8f z = {};
    v8f s0 = WMMA_BF16(ka00, qb0, z);  s0 = WMMA_BF16(ka01, qb1, s0);
    v8f s1 = WMMA_BF16(ka10, qb0, z);  s1 = WMMA_BF16(ka11, qb1, s1);

    // scale + additive mask (key index = kt [+16] + r + 8*half)
    v4f mk0a = *reinterpret_cast<const v4f*>(mrow + kt + half * 8);
    v4f mk0b = *reinterpret_cast<const v4f*>(mrow + kt + half * 8 + 4);
    v4f mk1a = *reinterpret_cast<const v4f*>(mrow + kt + 16 + half * 8);
    v4f mk1b = *reinterpret_cast<const v4f*>(mrow + kt + 16 + half * 8 + 4);
    float sv0[8], sv1[8];
#pragma unroll
    for (int r = 0; r < 4; ++r) {
      sv0[r]     = s0[r]     * 0.125f + mk0a[r];
      sv0[r + 4] = s0[r + 4] * 0.125f + mk0b[r];
      sv1[r]     = s1[r]     * 0.125f + mk1a[r];
      sv1[r + 4] = s1[r + 4] * 0.125f + mk1b[r];
    }

    // online softmax (per q column; merge the two key-halves via shfl_xor 16)
    float tm = -1e30f;
#pragma unroll
    for (int r = 0; r < 8; ++r) tm = fmaxf(tm, fmaxf(sv0[r], sv1[r]));
    tm = fmaxf(tm, __shfl_xor(tm, 16, 32));
    const float newm = fmaxf(run_m, tm);
    const float corr = __expf(run_m - newm);
    run_m = newm;

    float p0[8], p1[8], lsum = 0.f;
#pragma unroll
    for (int r = 0; r < 8; ++r) {
      p0[r] = __expf(sv0[r] - newm);
      p1[r] = __expf(sv1[r] - newm);
      lsum += p0[r] + p1[r];
    }
    run_l = run_l * corr + lsum;

#pragma unroll
    for (int t = 0; t < 4; ++t)
#pragma unroll
      for (int e = 0; e < 8; ++e) ctx[t][e] *= corr;

    // p^T B-fragment: S0 tile supplies k=0..7(+8*half), S1 supplies k=16..23(+8*half)
    v16bf pb;
#pragma unroll
    for (int i = 0; i < 8; ++i) { pb[i] = f2bf(p0[i]); pb[i + 8] = f2bf(p1[i]); }

    // ctx^T += V^T(16d x 32k) x p^T(32k x 16q) over the 4 d-tiles
#pragma unroll
    for (int t = 0; t < 4; ++t) {
      v16bf va = load_frag(Vh + (size_t)(t * 16) * SS + kt, SS);
      ctx[t] = WMMA_BF16(va, pb, ctx[t]);
    }
  }

  const float l = run_l + __shfl_xor(run_l, 16, 32);
  const float rinv = 1.0f / l;

  // ctx^T C-layout: lane column n = query, rows d = t*16 + e + 8*half
  const int s = q0 + n;
  float* orow = out + ((size_t)b * SS + s) * HDIM + hh * HD;
#pragma unroll
  for (int t = 0; t < 4; ++t) {
    const int d0 = t * 16 + half * 8;
    v4f o0, o1;
#pragma unroll
    for (int r = 0; r < 4; ++r) { o0[r] = ctx[t][r] * rinv; o1[r] = ctx[t][r + 4] * rinv; }
    *reinterpret_cast<v4f*>(orow + d0)     = o0;
    *reinterpret_cast<v4f*>(orow + d0 + 4) = o1;
  }
}

// ---- Host launch ------------------------------------------------------------
extern "C" void kernel_launch(void* const* d_in, const int* in_sizes, int n_in,
                              void* d_out, int out_size, void* d_ws, size_t ws_size,
                              hipStream_t stream) {
  (void)in_sizes; (void)n_in; (void)out_size; (void)ws_size;
  const float* input0 = (const float*)d_in[0];
  const float* mask   = (const float*)d_in[1];
  const float* Wq = (const float*)d_in[2];
  const float* bq = (const float*)d_in[3];
  const float* Wk = (const float*)d_in[4];
  const float* bk = (const float*)d_in[5];
  const float* Wv = (const float*)d_in[6];
  const float* bv = (const float*)d_in[7];
  float* out = (float*)d_out;

  char* ws = (char*)d_ws;
  __bf16* Xb  = (__bf16*)ws; ws += (size_t)MM * HDIM * 2;       // 16 MB
  __bf16* Wqb = (__bf16*)ws; ws += (size_t)HDIM * HDIM * 2;     // 2 MB
  __bf16* Wkb = (__bf16*)ws; ws += (size_t)HDIM * HDIM * 2;
  __bf16* Wvb = (__bf16*)ws; ws += (size_t)HDIM * HDIM * 2;
  __bf16* Qb  = (__bf16*)ws; ws += (size_t)MM * HDIM * 2;       // [B,NH,S,HD]
  __bf16* Kb  = (__bf16*)ws; ws += (size_t)MM * HDIM * 2;       // [B,NH,S,HD]
  __bf16* Vb  = (__bf16*)ws; ws += (size_t)MM * HDIM * 2;       // [B,NH,HD,S]

  cvt_f32_to_bf16<<<(MM * HDIM) / 1024, 256, 0, stream>>>(input0, Xb, MM * HDIM);
  cvt_f32_to_bf16<<<(HDIM * HDIM) / 1024, 256, 0, stream>>>(Wq, Wqb, HDIM * HDIM);
  cvt_f32_to_bf16<<<(HDIM * HDIM) / 1024, 256, 0, stream>>>(Wk, Wkb, HDIM * HDIM);
  cvt_f32_to_bf16<<<(HDIM * HDIM) / 1024, 256, 0, stream>>>(Wv, Wvb, HDIM * HDIM);

  dim3 gg(HDIM / 256, MM / 64);   // 4 x 128 blocks, 256 threads (8 waves)
  qkv_gemm<<<gg, 256, 0, stream>>>(Xb, Wqb, bq, Qb, 0);
  qkv_gemm<<<gg, 256, 0, stream>>>(Xb, Wkb, bk, Kb, 0);
  qkv_gemm<<<gg, 256, 0, stream>>>(Xb, Wvb, bv, Vb, 1);

  attn_fwd<<<dim3(SS / 64, NHEAD, BB), 128, 0, stream>>>(Qb, Kb, Vb, mask, out);
}
